// SAMPart3D_76055280877708
// MI455X (gfx1250) — matrix operand order repaired and checked by reading, not compiled, not run on hardware
//
#include <hip/hip_runtime.h>
#include <hip/hip_bf16.h>
#include <math.h>

// ---------------------------------------------------------------------------
// SAMPart3D contrastive-grouping loss, MI455X (gfx1250) WMMA implementation.
//   - all 384x384-class GEMMs (11 layers x 2 scales batched as M=16384) via
//     v_wmma_f32_16x16x32_f16, wave tile 32x64 (8 WMMAs / K-step, 4/3x the
//     FLOP-per-load of a 32x32 tile)
//   - per-chunk pairwise-distance Gram matrices via the same WMMA path
//   - compute-bound problem (~55 GFLOP vs ~16MB working set, L2-resident)
// ---------------------------------------------------------------------------

typedef __attribute__((ext_vector_type(16))) _Float16 v16h;
typedef __attribute__((ext_vector_type(8)))  _Float16 v8h;
typedef __attribute__((ext_vector_type(8)))  float    v8f;

union AFrag { v16h v; v8h h2[2]; };

#define NPTS   8192
#define M2     16384      // both scales batched
#define D      384
#define KSTEPS 12         // 384 / 32

enum { EPI_RELU = 0, EPI_SCALEBIAS = 1, EPI_NONE = 2 };

// ---------------------------------------------------------------------------
// Generic WMMA GEMM:  C[M2,384] = epi( A[M2,384] @ Wt^T )
//   Wt is [N=384, K=384] f16 row-major (pre-transposed weights).
//   Block tile 128(M) x 128(N); 8 waves (4Mx2N); each wave: 32x64
//   (2x4 WMMA tiles -> 8 WMMAs per K-step vs 12 b128 loads).
// ---------------------------------------------------------------------------
template <int EPI>
__global__ __launch_bounds__(256) void gemm_wmma(
    const _Float16* __restrict__ A,
    const _Float16* __restrict__ Wt,
    _Float16* __restrict__ C,
    const float* __restrict__ wlast,   // f32 last row of W_in (len 384), EPI_SCALEBIAS
    const float* __restrict__ srow)    // f32 per-row scale (len M2), EPI_SCALEBIAS
{
    const int w    = threadIdx.x >> 5;
    const int lane = threadIdx.x & 31;
    const int lhalf = lane >> 4;
    const int l15   = lane & 15;

    const int m0 = blockIdx.x * 128 + (w & 3) * 32;   // wave M base (2 tiles)
    const int n0 = blockIdx.y * 128 + (w >> 2) * 64;  // wave N base (4 tiles)

    v8f acc[2][4] = {};

    for (int ks = 0; ks < KSTEPS; ++ks) {
        const int kb = ks * 32;
        v16h a[2], b[4];
        #pragma unroll
        for (int mt = 0; mt < 2; ++mt) {
            const _Float16* p = A + (size_t)(m0 + mt * 16 + l15) * D + kb + lhalf * 16;
            AFrag u;
            u.h2[0] = *(const v8h*)(p);
            u.h2[1] = *(const v8h*)(p + 8);
            a[mt] = u.v;
        }
        #pragma unroll
        for (int nt = 0; nt < 4; ++nt) {
            const _Float16* p = Wt + (size_t)(n0 + nt * 16 + l15) * D + kb + lhalf * 16;
            b[nt] = *(const v16h*)(p);
        }
        if (ks + 1 < KSTEPS) {  // hint next K-step's weight line (global_prefetch)
            __builtin_prefetch(Wt + (size_t)(n0 + l15) * D + kb + 32, 0, 3);
        }
        #pragma unroll
        for (int mt = 0; mt < 2; ++mt)
            #pragma unroll
            for (int nt = 0; nt < 4; ++nt)
                acc[mt][nt] = __builtin_amdgcn_wmma_f32_16x16x32_f16(
                    false, a[mt], false, b[nt], (short)0, acc[mt][nt], false, false);
    }

    // epilogue + store (C layout: lane l15 = col, VGPR v + 8*lhalf = row)
    #pragma unroll
    for (int mt = 0; mt < 2; ++mt) {
        #pragma unroll
        for (int nt = 0; nt < 4; ++nt) {
            const int col = n0 + nt * 16 + l15;
            #pragma unroll
            for (int v = 0; v < 8; ++v) {
                const int row = m0 + mt * 16 + lhalf * 8 + v;
                float x = acc[mt][nt][v];
                if constexpr (EPI == EPI_SCALEBIAS) x += srow[row] * wlast[col];
                if constexpr (EPI != EPI_NONE)      x = fmaxf(x, 0.0f);
                C[(size_t)row * D + col] = (_Float16)x;
            }
        }
    }
}

// ---------------------------------------------------------------------------
// Pairwise distance + masked loss. One block = 64x256 slab of one chunk's
// Gram matrix, two passes (normal-scale emb rows, enlarged-scale emb rows).
// ---------------------------------------------------------------------------
__global__ __launch_bounds__(256) void pdist_loss(
    const _Float16* __restrict__ emb,   // [16384, 384] f16
    const float* __restrict__ sq,       // [16384] squared norms
    const int* __restrict__ labels,     // [8192]
    float* __restrict__ accum)          // {sum_pos1, sum_pos2, sum_neg, n_pos, n_neg, n_blk}
{
    __shared__ int   s_lab[256];
    __shared__ float s_sq[2][256];

    const int c    = blockIdx.x >> 2;   // chunk 0..31
    const int slab = blockIdx.x & 3;    // 64-row slab
    const int t    = threadIdx.x;
    const int base[2] = { c * 256, NPTS + c * 256 };

    s_lab[t]   = labels[base[0] + t];
    s_sq[0][t] = sq[base[0] + t];
    s_sq[1][t] = sq[base[1] + t];
    __syncthreads();

    const int w = t >> 5, lane = t & 31;
    const int lhalf = lane >> 4, l15 = lane & 15;
    const int mrow  = slab * 64 + (w & 3) * 16;  // local row base (16 rows)
    const int nbase = (w >> 2) * 128;            // local col base (8 tiles)

    float vals[6] = {0.f, 0.f, 0.f, 0.f, 0.f, 0.f};

    for (int pass = 0; pass < 2; ++pass) {
        const int rb = base[pass];
        v8f acc[8] = {};
        for (int ks = 0; ks < KSTEPS; ++ks) {
            const int kb = ks * 32;
            const _Float16* pa = emb + (size_t)(rb + mrow + l15) * D + kb + lhalf * 16;
            AFrag ua;
            ua.h2[0] = *(const v8h*)(pa);
            ua.h2[1] = *(const v8h*)(pa + 8);
            const v16h a = ua.v;
            #pragma unroll
            for (int nt = 0; nt < 8; ++nt) {
                const _Float16* pb =
                    emb + (size_t)(rb + nbase + nt * 16 + l15) * D + kb + lhalf * 16;
                const v16h b = *(const v16h*)pb;
                acc[nt] = __builtin_amdgcn_wmma_f32_16x16x32_f16(
                    false, a, false, b, (short)0, acc[nt], false, false);
            }
        }
        #pragma unroll
        for (int nt = 0; nt < 8; ++nt) {
            const int sc = nbase + nt * 16 + l15;
            const int lj = s_lab[sc];
            const float sqj = s_sq[pass][sc];
            #pragma unroll
            for (int v = 0; v < 8; ++v) {
                const int sl = mrow + lhalf * 8 + v;
                const int li = s_lab[sl];
                const float d2 = s_sq[pass][sl] + sqj - 2.0f * acc[nt][v];
                const float d  = sqrtf(fmaxf(d2, 1e-12f));
                const bool blk = (sl <= sc) && (li != -1) && (lj != -1);
                const bool eq  = (li == lj);
                if (pass == 0) {
                    if (blk) {
                        vals[5] += 1.0f;
                        if (eq && (sl != sc)) { vals[3] += 1.0f; vals[0] += d; }
                        if (!eq)              { vals[4] += 1.0f; vals[2] += fmaxf(1.0f - d, 0.0f); }
                    }
                } else {
                    if (blk && eq && (sl != sc)) vals[1] += d;
                }
            }
        }
    }

    #pragma unroll
    for (int i = 0; i < 6; ++i)
        for (int off = 16; off > 0; off >>= 1)
            vals[i] += __shfl_down(vals[i], off, 32);
    if (lane == 0) {
        #pragma unroll
        for (int i = 0; i < 6; ++i) atomicAdd(&accum[i], vals[i]);
    }
}

// ---------------------------------------------------------------------------
// Small helper kernels
// ---------------------------------------------------------------------------
__global__ void zero_acc(float* a) { if (threadIdx.x < 8) a[threadIdx.x] = 0.0f; }

__global__ void build_scales(const float* __restrict__ scale,
                             const float* __restrict__ rand_u,
                             float* __restrict__ out)
{
    int r = blockIdx.x * blockDim.x + threadIdx.x;
    if (r >= M2) return;
    if (r < NPTS) out[r] = scale[r];
    else {
        float s = scale[r - NPTS];
        out[r] = s + fmaxf(2.0f - s, 0.0f) * rand_u[0];
    }
}

__global__ void build_x(const float* __restrict__ feat, _Float16* __restrict__ X)
{
    size_t idx = (size_t)blockIdx.x * blockDim.x + threadIdx.x;
    if (idx >= (size_t)M2 * D) return;
    size_t row = idx / D, col = idx % D;
    X[idx] = (_Float16)feat[(row & (NPTS - 1)) * D + col];
}

// transpose + f32->f16: dst[n*K + k] = src[k*stride + n]
__global__ void convert_wt(const float* __restrict__ src, _Float16* __restrict__ dst,
                           int K, int Nn, int stride)
{
    size_t idx = (size_t)blockIdx.x * blockDim.x + threadIdx.x;
    if (idx >= (size_t)K * Nn) return;
    int k = (int)(idx % K), n = (int)(idx / K);
    dst[(size_t)n * K + k] = (_Float16)src[(size_t)k * stride + n];
}

// pos-net input layer: K=7 (6 features + scale), pure VALU
__global__ void pos_in_layer(const float* __restrict__ org,
                             const float* __restrict__ Win,   // [7,384] f32
                             const float* __restrict__ scales,
                             _Float16* __restrict__ out)
{
    size_t idx = (size_t)blockIdx.x * blockDim.x + threadIdx.x;
    if (idx >= (size_t)M2 * D) return;
    int row = (int)(idx / D), n = (int)(idx % D);
    const float* o = org + (size_t)(row & (NPTS - 1)) * 6;
    float h = scales[row] * Win[6 * D + n];
    #pragma unroll
    for (int k = 0; k < 6; ++k) h += o[k] * Win[k * D + n];
    out[idx] = (_Float16)fmaxf(h, 0.0f);
}

// normalize inst & pos embeddings, add, store f16 emb + f32 squared norm.
// one wave per row, 8 rows per block.
__global__ __launch_bounds__(256) void normalize_add(
    const _Float16* __restrict__ inst, const _Float16* __restrict__ pos,
    _Float16* __restrict__ emb, float* __restrict__ sq)
{
    int row  = blockIdx.x * 8 + (threadIdx.x >> 5);
    int lane = threadIdx.x & 31;
    const _Float16* pi = inst + (size_t)row * D;
    const _Float16* pp = pos  + (size_t)row * D;
    float vi[12], vp[12], ssi = 0.f, ssp = 0.f;
    #pragma unroll
    for (int j = 0; j < 12; ++j) {
        vi[j] = (float)pi[lane + 32 * j];
        vp[j] = (float)pp[lane + 32 * j];
        ssi += vi[j] * vi[j];
        ssp += vp[j] * vp[j];
    }
    #pragma unroll
    for (int off = 16; off > 0; off >>= 1) {
        ssi += __shfl_xor(ssi, off, 32);
        ssp += __shfl_xor(ssp, off, 32);
    }
    float di = 1.0f / (sqrtf(ssi) + 1e-5f);
    float dp = 1.0f / (sqrtf(ssp) + 1e-5f);
    float sse = 0.f;
    #pragma unroll
    for (int j = 0; j < 12; ++j) {
        float e = vi[j] * di + vp[j] * dp;
        emb[(size_t)row * D + lane + 32 * j] = (_Float16)e;
        sse += e * e;
    }
    #pragma unroll
    for (int off = 16; off > 0; off >>= 1) sse += __shfl_xor(sse, off, 32);
    if (lane == 0) sq[row] = sse;
}

__global__ void finalize(const float* __restrict__ a, float* __restrict__ out)
{
    float np = fmaxf(a[3], 1.0f), nn = fmaxf(a[4], 1.0f), nb = fmaxf(a[5], 1.0f);
    float l1 = a[0] / np, l2 = a[1] / np, l3 = a[2] / nn;
    float wp = a[3] / nb, wn = a[4] / nb;
    out[0] = l1 * wp + l2 * wp + l3 * wn;
}

// ---------------------------------------------------------------------------
extern "C" void kernel_launch(void* const* d_in, const int* in_sizes, int n_in,
                              void* d_out, int out_size, void* d_ws, size_t ws_size,
                              hipStream_t stream)
{
    const float* feat   = (const float*)d_in[0];   // [8192,384]
    const float* org    = (const float*)d_in[1];   // [8192,6]
    const float* scale  = (const float*)d_in[2];   // [8192,1]
    const int*   labels = (const int*)d_in[3];     // [8192]
    const float* rand_u = (const float*)d_in[4];   // [1]
    const float* iWin   = (const float*)d_in[6];   // [385,384]
    const float* iWhid  = (const float*)d_in[7];   // [5,384,384]
    const float* iWout  = (const float*)d_in[8];   // [384,384]
    const float* pWin   = (const float*)d_in[9];   // [7,384]
    const float* pWhid  = (const float*)d_in[10];  // [3,384,384]
    const float* pWout  = (const float*)d_in[11];  // [384,384]

    char* ws = (char*)d_ws;
    size_t off = 0;
    const size_t WMAT  = (size_t)D * D * sizeof(_Float16);     // 294912 B
    const size_t BIGBUF = (size_t)M2 * D * sizeof(_Float16);   // ~12.6 MB

    float* acc    = (float*)(ws + off); off += 256;
    float* scales = (float*)(ws + off); off += (size_t)M2 * 4;
    float* sq     = (float*)(ws + off); off += (size_t)M2 * 4;
    _Float16* w_iin  = (_Float16*)(ws + off); off += WMAT;
    _Float16* w_ihid[5];
    for (int l = 0; l < 5; ++l) { w_ihid[l] = (_Float16*)(ws + off); off += WMAT; }
    _Float16* w_iout = (_Float16*)(ws + off); off += WMAT;
    _Float16* w_phid[3];
    for (int l = 0; l < 3; ++l) { w_phid[l] = (_Float16*)(ws + off); off += WMAT; }
    _Float16* w_pout = (_Float16*)(ws + off); off += WMAT;
    _Float16* buf0 = (_Float16*)(ws + off); off += BIGBUF;
    _Float16* buf1 = (_Float16*)(ws + off); off += BIGBUF;
    _Float16* buf2 = (_Float16*)(ws + off); off += BIGBUF;

    const int T = 256;
    const int elemBlocks = (int)(((size_t)M2 * D + T - 1) / T);
    const int wtBlocks   = (int)(((size_t)D * D + T - 1) / T);
    dim3 gemmGrid(M2 / 128, D / 128);   // (128, 3)

    zero_acc<<<1, 32, 0, stream>>>(acc);
    build_scales<<<(M2 + T - 1) / T, T, 0, stream>>>(scale, rand_u, scales);
    build_x<<<elemBlocks, T, 0, stream>>>(feat, buf0);

    // weight convert + transpose (all [384,384] with row stride 384)
    convert_wt<<<wtBlocks, T, 0, stream>>>(iWin, w_iin, D, D, D);  // first 384 rows
    for (int l = 0; l < 5; ++l)
        convert_wt<<<wtBlocks, T, 0, stream>>>(iWhid + (size_t)l * D * D, w_ihid[l], D, D, D);
    convert_wt<<<wtBlocks, T, 0, stream>>>(iWout, w_iout, D, D, D);
    for (int l = 0; l < 3; ++l)
        convert_wt<<<wtBlocks, T, 0, stream>>>(pWhid + (size_t)l * D * D, w_phid[l], D, D, D);
    convert_wt<<<wtBlocks, T, 0, stream>>>(pWout, w_pout, D, D, D);

    // ---- instance net (7 WMMA layers), ping-pong buf0 <-> buf1 ----
    gemm_wmma<EPI_SCALEBIAS><<<gemmGrid, T, 0, stream>>>(buf0, w_iin, buf1,
                                                         iWin + (size_t)D * D, scales);
    gemm_wmma<EPI_RELU><<<gemmGrid, T, 0, stream>>>(buf1, w_ihid[0], buf0, nullptr, nullptr);
    gemm_wmma<EPI_RELU><<<gemmGrid, T, 0, stream>>>(buf0, w_ihid[1], buf1, nullptr, nullptr);
    gemm_wmma<EPI_RELU><<<gemmGrid, T, 0, stream>>>(buf1, w_ihid[2], buf0, nullptr, nullptr);
    gemm_wmma<EPI_RELU><<<gemmGrid, T, 0, stream>>>(buf0, w_ihid[3], buf1, nullptr, nullptr);
    gemm_wmma<EPI_RELU><<<gemmGrid, T, 0, stream>>>(buf1, w_ihid[4], buf0, nullptr, nullptr);
    gemm_wmma<EPI_NONE><<<gemmGrid, T, 0, stream>>>(buf0, w_iout, buf1, nullptr, nullptr);
    // inst_out = buf1

    // ---- pos net: VALU input layer, then 4 WMMA layers ----
    pos_in_layer<<<elemBlocks, T, 0, stream>>>(org, pWin, scales, buf0);
    gemm_wmma<EPI_RELU><<<gemmGrid, T, 0, stream>>>(buf0, w_phid[0], buf2, nullptr, nullptr);
    gemm_wmma<EPI_RELU><<<gemmGrid, T, 0, stream>>>(buf2, w_phid[1], buf0, nullptr, nullptr);
    gemm_wmma<EPI_RELU><<<gemmGrid, T, 0, stream>>>(buf0, w_phid[2], buf2, nullptr, nullptr);
    gemm_wmma<EPI_NONE><<<gemmGrid, T, 0, stream>>>(buf2, w_pout, buf0, nullptr, nullptr);
    // pos_out = buf0

    normalize_add<<<M2 / 8, T, 0, stream>>>(buf1, buf0, buf2, sq);
    pdist_loss<<<128, T, 0, stream>>>(buf2, sq, labels, acc);
    finalize<<<1, 1, 0, stream>>>(acc, (float*)d_out);
}
